// FullAttention_22265110463187
// MI455X (gfx1250) — matrix operand order repaired and checked by reading, not compiled, not run on hardware
//
#include <hip/hip_runtime.h>

typedef __attribute__((ext_vector_type(16))) __bf16 v16bf;
typedef __attribute__((ext_vector_type(8)))  __bf16 v8bf;
typedef __attribute__((ext_vector_type(8)))  float  v8f;

#define B_   4
#define L_   2048
#define H_   16
#define E_   64
#define S_   L_

#define BLOCK_Q 128        // query rows per workgroup (8 waves x 16 rows)
#define KT      64         // kv chunk per outer iteration (2 x 32-K sub-chunks)

#define TSTR 72            // LDS tile row stride (bf16): 144B rows (128B data + 16B pad)
#define TBUF (64 * TSTR)   // one 64-row tile buffer (elements)
#define PSTR 40            // P_lds row stride (bf16)

#define NEGINF (-1.0e30f)
#define LOG2E  1.44269504088896340736f

__device__ __forceinline__ v8bf cvt8(const float* p) {
    float4 a = *(const float4*)p;
    float4 b = *(const float4*)(p + 4);
    v8bf r;
    r[0] = (__bf16)a.x; r[1] = (__bf16)a.y; r[2] = (__bf16)a.z; r[3] = (__bf16)a.w;
    r[4] = (__bf16)b.x; r[5] = (__bf16)b.y; r[6] = (__bf16)b.z; r[7] = (__bf16)b.w;
    return r;
}

__device__ __forceinline__ v16bf cat16(v8bf lo, v8bf hi) {
    return __builtin_shufflevector(lo, hi, 0,1,2,3,4,5,6,7,8,9,10,11,12,13,14,15);
}

// max-reduce across each 16-lane row via DPP ROW_XMASK (VALU only)
__device__ __forceinline__ float dppmax16(float x) {
    float y;
    y = __int_as_float(__builtin_amdgcn_mov_dpp(__float_as_int(x), 0x161, 0xf, 0xf, true)); x = fmaxf(x, y);
    y = __int_as_float(__builtin_amdgcn_mov_dpp(__float_as_int(x), 0x162, 0xf, 0xf, true)); x = fmaxf(x, y);
    y = __int_as_float(__builtin_amdgcn_mov_dpp(__float_as_int(x), 0x164, 0xf, 0xf, true)); x = fmaxf(x, y);
    y = __int_as_float(__builtin_amdgcn_mov_dpp(__float_as_int(x), 0x168, 0xf, 0xf, true)); x = fmaxf(x, y);
    return x;
}

// CDNA5 async DMA: 16B global -> LDS, tracked by ASYNCcnt (CDNA5 ISA)
__device__ __forceinline__ void async_ld16(unsigned lds_off, const void* gaddr) {
    asm volatile("global_load_async_to_lds_b128 %0, %1, off"
                 :: "v"(lds_off), "v"(gaddr) : "memory");
}
__device__ __forceinline__ void wait_async0() {
    asm volatile("s_wait_asynccnt 0" ::: "memory");
}

// ---------------- pre-pass 1: K fp32 [B,S,H,E] -> bf16 [B,H,S,E] ----------------
__global__ __launch_bounds__(256, 1)
void cvtK(const float* __restrict__ K, __bf16* __restrict__ Kb) {
    const int gid   = blockIdx.x * 256 + threadIdx.x;
    const int rowid = gid >> 2;              // (b*S + s)*H + h
    const int seg   = (gid & 3) * 16;
    const int h  = rowid & 15;
    const int s  = (rowid >> 4) & (S_ - 1);
    const int b  = rowid >> 15;
    const float* src = K  + (((size_t)b * S_ + s) * H_ + h) * E_ + seg;
    __bf16*      dst = Kb + (((size_t)b * H_ + h) * S_ + s) * E_ + seg;
    *(v8bf*)dst       = cvt8(src);
    *(v8bf*)(dst + 8) = cvt8(src + 8);
}

// ------------- pre-pass 2: V fp32 [B,S,H,E] -> bf16 transposed [B,H,E,S] -------------
__global__ __launch_bounds__(256, 1)
void cvtV(const float* __restrict__ V, __bf16* __restrict__ Vt) {
    __shared__ __align__(16) __bf16 T[64 * TSTR];
    const int tid = threadIdx.x;
    const int bx  = blockIdx.x;
    const int s0  = (bx & 31) * 64;          // S_/64 == 32
    const int h   = (bx >> 5) & 15;
    const int b   = bx >> 9;

    const int i = tid >> 2, c = (tid & 3) * 16;
    const float* src = V + (((size_t)b * S_ + s0 + i) * H_ + h) * E_ + c;
    *(v8bf*)&T[i * TSTR + c]     = cvt8(src);
    *(v8bf*)&T[i * TSTR + c + 8] = cvt8(src + 8);
    __syncthreads();

    const int e = tid >> 2, oc = (tid & 3) * 16;
    v8bf o0, o1;
    #pragma unroll
    for (int j = 0; j < 8; ++j) {
        o0[j] = T[(oc + j) * TSTR + e];
        o1[j] = T[(oc + 8 + j) * TSTR + e];
    }
    __bf16* dst = Vt + (((size_t)b * H_ + h) * E_ + e) * S_ + s0 + oc;
    *(v8bf*)dst       = o0;
    *(v8bf*)(dst + 8) = o1;
}

// ---------------- main: causal flash attention, bf16 WMMA ----------------
__global__ __launch_bounds__(256, 1)
void fa_causal_bf16wmma(const float* __restrict__ Qg,
                        const __bf16* __restrict__ Kb,
                        const __bf16* __restrict__ Vtb,
                        float* __restrict__ Og)
{
    __shared__ __align__(16) __bf16 K_lds[2 * TBUF];   // double-buffered K tile (kv x e)
    __shared__ __align__(16) __bf16 Vt_lds[2 * TBUF];  // double-buffered V^T tile (e x kv)
    __shared__ __align__(16) __bf16 P_lds[8 * 16 * PSTR];

    const int tid  = threadIdx.x;
    const int wid  = tid >> 5;
    const int lane = tid & 31;
    const int ln16 = lane & 15;
    const int hi   = (lane >> 4) & 1;

    const int bx   = blockIdx.x;
    const int qblk = bx & 15;
    const int h    = (bx >> 4) & 15;
    const int b    = bx >> 8;

    const int q0     = qblk * BLOCK_Q;
    const int wqbase = q0 + wid * 16;
    const int wq_u   = __builtin_amdgcn_readfirstlane(wqbase);

    const int abase = hi ? 8 : 0;
    const int bbase = hi ? 16 : 0;
    const int qrow  = wqbase + ln16;
    const float* qptr = Qg + (((size_t)b * L_ + qrow) * H_ + h) * E_;

    v16bf qa[2];
    #pragma unroll
    for (int c = 0; c < 2; ++c)
        qa[c] = cat16(cvt8(qptr + c * 32 + bbase), cvt8(qptr + c * 32 + bbase + 8));

    v16bf onesb;
    #pragma unroll
    for (int i = 0; i < 16; ++i) onesb[i] = (__bf16)1.0f;

    v8f acc[4];
    #pragma unroll
    for (int t = 0; t < 4; ++t)
        acc[t] = (v8f){0.f,0.f,0.f,0.f,0.f,0.f,0.f,0.f};
    v8f lacc = (v8f){0.f,0.f,0.f,0.f,0.f,0.f,0.f,0.f};
    float mrow[8];
    #pragma unroll
    for (int r = 0; r < 8; ++r) mrow[r] = NEGINF;

    const float scl = 0.125f * LOG2E;

    // bf16 source bases for this (b,h)
    const char* kgbase = (const char*)(Kb  + ((size_t)(b * H_ + h)) * S_ * E_);
    const char* vgbase = (const char*)(Vtb + ((size_t)(b * H_ + h)) * E_ * S_);
    const unsigned kldsb = (unsigned)(uintptr_t)&K_lds[0];
    const unsigned vldsb = (unsigned)(uintptr_t)&Vt_lds[0];

    // each thread DMAs 2x16B of K and 2x16B of V^T per chunk (512 x 16B chunks each)
    auto issue = [&](int buf, int kv0) {
        #pragma unroll
        for (int j = 0; j < 2; ++j) {
            const int c   = tid + 256 * j;
            const int row = c >> 3;            // kv row (K) / e row (V^T)
            const int col = (c & 7) * 16;      // byte offset within 128B row
            async_ld16(kldsb + (unsigned)(buf * 2 * TBUF + row * 144 + col),
                       kgbase + (size_t)(kv0 + row) * (E_ * 2) + col);
            async_ld16(vldsb + (unsigned)(buf * 2 * TBUF + row * 144 + col),
                       vgbase + (size_t)row * (S_ * 2) + (size_t)kv0 * 2 + col);
        }
    };

    const int kv_end  = q0 + BLOCK_Q;
    const int nchunks = kv_end / KT;

    issue(0, 0);

    for (int ci = 0; ci < nchunks; ++ci) {
        const int kv0 = ci * KT;
        const int buf = ci & 1;
        wait_async0();            // own DMAs for buf done
        __syncthreads();          // everyone's DMAs visible; prev compute finished
        if (ci + 1 < nchunks) issue(buf ^ 1, kv0 + KT);   // overlap DMA with compute

        if (kv0 <= wq_u + 15) {                 // uniform scalar branch
            const __bf16* Kc = &K_lds[buf * TBUF];
            const __bf16* Vc = &Vt_lds[buf * TBUF];
            const int  nsub     = (kv0 + 32 <= wq_u + 15) ? 2 : 1;
            const bool needmask = (kv0 + nsub * 32 - 1) > wq_u;

            v8f s2[4];
            auto qk_tile = [&](int t) {
                v8f s = (v8f){0.f,0.f,0.f,0.f,0.f,0.f,0.f,0.f};
                #pragma unroll
                for (int c = 0; c < 2; ++c) {
                    const __bf16* kp = &Kc[(t * 16 + ln16) * TSTR + c * 32 + bbase];
                    v16bf kb = cat16(*(const v8bf*)kp, *(const v8bf*)(kp + 8));
                    s = __builtin_amdgcn_wmma_f32_16x16x32_bf16(
                            false, qa[c], false, kb, (short)0, s, false, false);
                }
                #pragma unroll
                for (int r = 0; r < 8; ++r) s[r] *= scl;
                s2[t] = s;
            };
            qk_tile(0); qk_tile(1);
            if (nsub == 2) { qk_tile(2); qk_tile(3); }

            if (needmask) {
                auto mask_tile = [&](int t) {
                    #pragma unroll
                    for (int r = 0; r < 8; ++r) {
                        int q = wqbase + r + (hi ? 8 : 0);
                        int k = kv0 + t * 16 + ln16;
                        if (k > q) s2[t][r] = NEGINF;
                    }
                };
                mask_tile(0); mask_tile(1);
                if (nsub == 2) { mask_tile(2); mask_tile(3); }
            }

            float mnew[8], alpha[8];
            #pragma unroll
            for (int r = 0; r < 8; ++r) {
                float mx = fmaxf(s2[0][r], s2[1][r]);
                if (nsub == 2) mx = fmaxf(mx, fmaxf(s2[2][r], s2[3][r]));
                mx = dppmax16(mx);
                mnew[r]  = fmaxf(mrow[r], mx);
                alpha[r] = exp2f(mrow[r] - mnew[r]);
                mrow[r]  = mnew[r];
            }
            #pragma unroll
            for (int t = 0; t < 4; ++t)
                #pragma unroll
                for (int r = 0; r < 8; ++r)
                    acc[t][r] *= alpha[r];
            #pragma unroll
            for (int r = 0; r < 8; ++r) lacc[r] *= alpha[r];

            auto exp_tile = [&](int t) {
                #pragma unroll
                for (int r = 0; r < 8; ++r)
                    s2[t][r] = exp2f(s2[t][r] - mnew[r]);
            };
            exp_tile(0); exp_tile(1);
            if (nsub == 2) { exp_tile(2); exp_tile(3); }

            __bf16* pl = &P_lds[wid * 16 * PSTR];
            auto pv_sub = [&](int s) {
                #pragma unroll
                for (int tt = 0; tt < 2; ++tt) {
                    const int t = s * 2 + tt;
                    #pragma unroll
                    for (int r = 0; r < 8; ++r)
                        pl[(r + (hi ? 8 : 0)) * PSTR + tt * 16 + ln16] = (__bf16)s2[t][r];
                }
                const __bf16* pp = pl + ln16 * PSTR + bbase;
                v16bf pa = cat16(*(const v8bf*)pp, *(const v8bf*)(pp + 8));

                lacc = __builtin_amdgcn_wmma_f32_16x16x32_bf16(
                           false, pa, false, onesb, (short)0, lacc, false, false);

                #pragma unroll
                for (int t16 = 0; t16 < 4; ++t16) {
                    const __bf16* vp = &Vc[(t16 * 16 + ln16) * TSTR + s * 32 + bbase];
                    v16bf vbf = cat16(*(const v8bf*)vp, *(const v8bf*)(vp + 8));
                    acc[t16] = __builtin_amdgcn_wmma_f32_16x16x32_bf16(
                                   false, pa, false, vbf, (short)0, acc[t16], false, false);
                }
            };
            pv_sub(0);
            if (nsub == 2) pv_sub(1);
        }
        __syncthreads();   // compute done before next iteration overwrites buffers
    }

    float invl[8];
    #pragma unroll
    for (int r = 0; r < 8; ++r) invl[r] = 1.0f / lacc[r];

    #pragma unroll
    for (int t = 0; t < 4; ++t)
        #pragma unroll
        for (int r = 0; r < 8; ++r) {
            int q = wqbase + r + (hi ? 8 : 0);
            int e = t * 16 + ln16;
            Og[(((size_t)b * L_ + q) * H_ + h) * E_ + e] = acc[t][r] * invl[r];
        }
}

extern "C" void kernel_launch(void* const* d_in, const int* in_sizes, int n_in,
                              void* d_out, int out_size, void* d_ws, size_t ws_size,
                              hipStream_t stream) {
    const float* q = (const float*)d_in[0];
    const float* k = (const float*)d_in[1];
    const float* v = (const float*)d_in[2];
    float* o = (float*)d_out;
    (void)in_sizes; (void)n_in; (void)out_size; (void)ws_size;

    // workspace: K bf16 [B,H,S,E] then V^T bf16 [B,H,E,S]  (2 x 16.8 MB)
    __bf16* Kb  = (__bf16*)d_ws;
    __bf16* Vtb = Kb + (size_t)B_ * H_ * S_ * E_;

    cvtK<<<dim3((B_ * S_ * H_ * 4) / 256), dim3(256), 0, stream>>>(k, Kb);
    cvtV<<<dim3(B_ * H_ * (S_ / 64)),      dim3(256), 0, stream>>>(v, Vtb);

    dim3 grid(B_ * H_ * (L_ / BLOCK_Q));   // 1024 workgroups
    fa_causal_bf16wmma<<<grid, dim3(256), 0, stream>>>(q, Kb, Vtb, o);
}
